// Attention_9680856285639
// MI455X (gfx1250) — matrix operand order repaired and checked by reading, not compiled, not run on hardware
//
#include <hip/hip_runtime.h>
#include <hip/hip_bf16.h>

// ---------------------------------------------------------------------------
// MHA block for MI455X (gfx1250): f16 WMMA (v_wmma_f32_16x16x32_f16) for all
// GEMMs, f32 accumulation, flash-style online softmax, wave32 layouts,
// Tensor Data Mover (tensor_load_to_lds) for double-buffered B-tile staging.
// ---------------------------------------------------------------------------

typedef __attribute__((ext_vector_type(16))) _Float16 v16h;
typedef __attribute__((ext_vector_type(8)))  float    v8f;
typedef __attribute__((ext_vector_type(4)))  unsigned int u32x4;
typedef __attribute__((ext_vector_type(8)))  int          i32x8;
typedef __attribute__((ext_vector_type(4)))  int          i32x4;

#define D_MODEL 2048
#define N_HEADS 16
#define D_HEAD  128
#define BATCH   2
#define SEQ     2048
#define BS      (BATCH * SEQ)

#define WMMA16(a, b, c) \
  __builtin_amdgcn_wmma_f32_16x16x32_f16(false, (a), false, (b), (short)0, (c), false, false)

// Load a v16h fragment from two 16-byte-aligned 8-half chunks.
__device__ __forceinline__ v16h frag_load(const _Float16* p0, const _Float16* p1) {
  union { v16h h; uint4 u[2]; } t;
  t.u[0] = *(const uint4*)p0;
  t.u[1] = *(const uint4*)p1;
  return t.h;
}

// ---------------------------------------------------------------------------
// TDM: async-load a 2D f16 tile (rows x cols, cols contiguous in memory with
// row stride `stride` elements) into LDS at byte offset lds_off, packed
// row-major (rows of `cols` halves).  Descriptor per CDNA5 ISA §8.3/8.4.
// 6-arg toolchain signature: (g0, g1, g2, g3, g4, cpol); groups 2..4 unused
// for 2D tiles (VADDR2/3 disabled) -> zeros.  Tracked with TENSORcnt.
// ---------------------------------------------------------------------------
__device__ __forceinline__ void tdm_load_tile_f16(const _Float16* gsrc,
                                                  unsigned lds_off,
                                                  unsigned rows, unsigned cols,
                                                  unsigned tensor_w, unsigned tensor_h,
                                                  unsigned stride) {
  unsigned long long ga = (unsigned long long)(size_t)gsrc;
  u32x4 g0;
  g0[0] = 1u;                                              // count=1, user desc
  g0[1] = lds_off;                                         // lds_addr (bytes)
  g0[2] = (unsigned)(ga & 0xffffffffu);                    // global_addr[31:0]
  g0[3] = (unsigned)((ga >> 32) & 0x01ffffffu) | (2u << 30); // addr[56:32], type=2
  i32x8 g1;
  g1[0] = (int)(1u << 16);                                 // data_size=1 (2 bytes)
  g1[1] = (int)((tensor_w & 0xffffu) << 16);               // tensor_dim0[15:0]
  g1[2] = (int)(((tensor_w >> 16) & 0xffffu) |
                ((tensor_h & 0xffffu) << 16));             // dim0 hi | dim1 lo
  g1[3] = (int)(((tensor_h >> 16) & 0xffffu) |
                ((cols & 0xffffu) << 16));                 // dim1 hi | tile_dim0
  g1[4] = (int)(rows & 0xffffu);                           // tile_dim1 (dim2=0)
  g1[5] = (int)stride;                                     // dim0_stride[31:0]
  g1[6] = 0;                                               // stride hi, dim1_stride lo
  g1[7] = 0;
  i32x4 z4 = {0, 0, 0, 0};
  i32x8 z8 = {0, 0, 0, 0, 0, 0, 0, 0};
  __builtin_amdgcn_tensor_load_to_lds(g0, g1, z4, z4, z8, 0);
}

// ===========================================================================
// Kernel 0a: elementwise f32 -> f16 cast (for x).
// ===========================================================================
__global__ __launch_bounds__(256)
void cast_kernel(const float* __restrict__ src, _Float16* __restrict__ dst) {
  size_t i = ((size_t)blockIdx.x * 256 + threadIdx.x) * 4;
  float4 v = *(const float4*)(src + i);
  dst[i + 0] = (_Float16)v.x;
  dst[i + 1] = (_Float16)v.y;
  dst[i + 2] = (_Float16)v.z;
  dst[i + 3] = (_Float16)v.w;
}

// ===========================================================================
// Kernel 0b: batched transpose + cast:  src[z][R][C] f32 -> dst[z][C][R] f16.
// 32x32 LDS tiles, 256 threads.
// ===========================================================================
__global__ __launch_bounds__(256)
void transpose_cast_kernel(const float* __restrict__ src, _Float16* __restrict__ dst,
                           int R, int C) {
  __shared__ _Float16 t[32][33];
  const float*  s = src + (size_t)blockIdx.z * R * C;
  _Float16*     d = dst + (size_t)blockIdx.z * R * C;
  int tx = threadIdx.x & 31, ty = threadIdx.x >> 5;
  int r0 = blockIdx.y * 32, c0 = blockIdx.x * 32;
#pragma unroll
  for (int i = 0; i < 32; i += 8)
    t[ty + i][tx] = (_Float16)s[(size_t)(r0 + ty + i) * C + c0 + tx];
  __syncthreads();
#pragma unroll
  for (int i = 0; i < 32; i += 8)
    d[(size_t)(c0 + ty + i) * R + r0 + tx] = t[tx][ty + i];
}

// ===========================================================================
// Kernel 1: QKV projection.  q/k/v = x @ W_{q,k,v} + b  (per head).
// A-fragments: direct 16B global loads from xh (f16).
// B-operands: pre-transposed WT[h][e][d]; 128x32 tiles DMA'd to LDS by the
// TDM, double buffered (3 tiles/step), s_wait_tensorcnt + barrier sync.
// Outputs: Qh,Kh as [b][h][s][e] f16;  Vt transposed as [b][h][e][s] f16.
// ===========================================================================
__global__ __launch_bounds__(256)
void qkv_kernel(const _Float16* __restrict__ xh,
                const _Float16* __restrict__ WqT,
                const _Float16* __restrict__ WkT,
                const _Float16* __restrict__ WvT,
                const float* __restrict__ bq,
                const float* __restrict__ bk,
                const float* __restrict__ bv,
                _Float16* __restrict__ Qh,
                _Float16* __restrict__ Kh,
                _Float16* __restrict__ Vt) {
  __shared__ _Float16 sW[2][3][128 * 32];   // [buf][matrix][n*32+k]

  const int tid  = threadIdx.x;
  const int wid  = tid >> 5;
  const int lane = tid & 31;
  const int r0   = blockIdx.x * 128;        // flat b*s row base
  const int h    = blockIdx.y;

  const _Float16* WT[3] = {WqT + (size_t)h * D_HEAD * D_MODEL,
                           WkT + (size_t)h * D_HEAD * D_MODEL,
                           WvT + (size_t)h * D_HEAD * D_MODEL};

  v8f accq[8] = {}, acck[8] = {}, accv[8] = {};

  const int klo = (lane < 16) ? 0 : 16;     // A-frag K base (contiguous k = 16h + s)
  const int kbl = (lane < 16) ? 0 : 16;     // B-frag K base
  const int mr  = r0 + wid * 16 + (lane & 15);

  // Prologue: DMA the first 3 W tiles into buffer 0.
  if (wid == 0) {
#pragma unroll
    for (int m = 0; m < 3; m++)
      tdm_load_tile_f16(WT[m], (unsigned)(size_t)&sW[0][m][0],
                        /*rows*/ D_HEAD, /*cols*/ 32,
                        /*tensor_w*/ D_MODEL, /*tensor_h*/ D_HEAD,
                        /*stride*/ D_MODEL);
  }

  const int NSTEP = D_MODEL / 32;
  for (int ks = 0; ks < NSTEP; ks++) {
    const int k0  = ks * 32;
    const int cur = ks & 1;
    if (wid == 0) {
      if (ks + 1 < NSTEP) {
        const int nxt = (ks + 1) & 1;
#pragma unroll
        for (int m = 0; m < 3; m++)
          tdm_load_tile_f16(WT[m] + (k0 + 32), (unsigned)(size_t)&sW[nxt][m][0],
                            D_HEAD, 32, D_MODEL, D_HEAD, D_MODEL);
        __builtin_amdgcn_s_wait_tensorcnt(3);   // current step's 3 tiles done
      } else {
        __builtin_amdgcn_s_wait_tensorcnt(0);
      }
    }
    __syncthreads();

    const _Float16* pa = xh + (size_t)mr * D_MODEL + k0 + klo;
    v16h a = frag_load(pa, pa + 8);
#pragma unroll
    for (int nt = 0; nt < 8; nt++) {
      int n = nt * 16 + (lane & 15);
      v16h bq_f = frag_load(&sW[cur][0][n * 32 + kbl], &sW[cur][0][n * 32 + kbl + 8]);
      accq[nt] = WMMA16(a, bq_f, accq[nt]);
      v16h bk_f = frag_load(&sW[cur][1][n * 32 + kbl], &sW[cur][1][n * 32 + kbl + 8]);
      acck[nt] = WMMA16(a, bk_f, acck[nt]);
      v16h bv_f = frag_load(&sW[cur][2][n * 32 + kbl], &sW[cur][2][n * 32 + kbl + 8]);
      accv[nt] = WMMA16(a, bv_f, accv[nt]);
    }
    __syncthreads();   // all waves done with buf[cur] before TDM refills it
  }

  // ---- epilogue: bias, f16 convert, scatter to Q/K ([b][h][s][e]) and Vt ----
  const int lh = (lane < 16) ? 0 : 8;
#pragma unroll
  for (int nt = 0; nt < 8; nt++) {
    int n = nt * 16 + (lane & 15);
    float biasq = bq[h * D_HEAD + n];
    float biask = bk[h * D_HEAD + n];
    float biasv = bv[h * D_HEAD + n];
#pragma unroll
    for (int v = 0; v < 8; v++) {
      int row = r0 + wid * 16 + v + lh;
      int b   = row / SEQ;
      int s   = row % SEQ;
      size_t qk = (((size_t)(b * N_HEADS + h)) * SEQ + s) * D_HEAD + n;
      Qh[qk] = (_Float16)(accq[nt][v] + biasq);
      Kh[qk] = (_Float16)(acck[nt][v] + biask);
      size_t vt = (((size_t)(b * N_HEADS + h)) * D_HEAD + n) * SEQ + s;
      Vt[vt] = (_Float16)(accv[nt][v] + biasv);
    }
  }
}

// ===========================================================================
// Kernel 2: causal flash attention.  One (b, h, 128 q-rows) per block.
// Each wave owns 16 q-rows; Q fragments live in registers across the KV
// sweep; K/V fragments come straight from global (L2-resident).
// P goes C-layout -> LDS -> A-layout for the PV WMMAs.
// Output Zh: [b][s][h][e] f16 (flat K-dim = 2048 for the out-proj GEMM).
// ===========================================================================
__global__ __launch_bounds__(256)
void flash_kernel(const _Float16* __restrict__ Qh,
                  const _Float16* __restrict__ Kh,
                  const _Float16* __restrict__ Vt,
                  _Float16* __restrict__ Zh) {
  __shared__ _Float16 sP[8][16 * 32];     // per-wave P tile [m][kk]

  const int tid  = threadIdx.x;
  const int wid  = tid >> 5;
  const int lane = tid & 31;
  const int h = blockIdx.y, b = blockIdx.z;
  const int qlo = blockIdx.x * 128 + wid * 16;

  const size_t headQK = ((size_t)(b * N_HEADS + h)) * SEQ;     // row base
  const size_t headV  = ((size_t)(b * N_HEADS + h)) * D_HEAD;  // e base

  const int klo = (lane < 16) ? 0 : 16;
  const int kbl = (lane < 16) ? 0 : 16;
  const int n   = lane & 15;
  const int lh  = (lane < 16) ? 0 : 8;

  // Q fragments (16 rows x 128 e) held in registers.
  v16h qf[4];
  {
    const int sq = qlo + n;
#pragma unroll
    for (int ec = 0; ec < 4; ec++) {
      const _Float16* p = Qh + (headQK + sq) * D_HEAD + ec * 32 + klo;
      qf[ec] = frag_load(p, p + 8);
    }
  }

  v8f z[8] = {};
  float mrow[8], lrow[8];
#pragma unroll
  for (int v = 0; v < 8; v++) { mrow[v] = -1e30f; lrow[v] = 0.f; }

  const float scale = 0.08838834764831845f;  // 1/sqrt(128)
  const int kmax = qlo + 16;                 // causal bound (wave-uniform)

  for (int kt = 0; kt < kmax; kt += 32) {
    // ---- scores: two 16x16 tiles over kv cols [kt, kt+32) ----
    v8f s0 = {}, s1 = {};
#pragma unroll
    for (int ec = 0; ec < 4; ec++) {
      const _Float16* pk0 = Kh + (headQK + kt + n) * D_HEAD + ec * 32 + kbl;
      s0 = WMMA16(qf[ec], frag_load(pk0, pk0 + 8), s0);
      const _Float16* pk1 = Kh + (headQK + kt + 16 + n) * D_HEAD + ec * 32 + kbl;
      s1 = WMMA16(qf[ec], frag_load(pk1, pk1 + 8), s1);
    }

    // ---- online softmax (row = vgpr v + lh, col = lane%16) ----
#pragma unroll
    for (int v = 0; v < 8; v++) {
      int rowq = qlo + v + lh;
      int c0 = kt + n, c1 = c0 + 16;
      float a0 = (c0 <= rowq) ? s0[v] * scale : -1e30f;
      float a1 = (c1 <= rowq) ? s1[v] * scale : -1e30f;
      float mx = fmaxf(a0, a1);
      mx = fmaxf(mx, __shfl_xor(mx, 1, 32));
      mx = fmaxf(mx, __shfl_xor(mx, 2, 32));
      mx = fmaxf(mx, __shfl_xor(mx, 4, 32));
      mx = fmaxf(mx, __shfl_xor(mx, 8, 32));
      float mnew  = fmaxf(mrow[v], mx);
      float alpha = __expf(mrow[v] - mnew);
      float p0 = __expf(a0 - mnew);
      float p1 = __expf(a1 - mnew);
      float rs = p0 + p1;
      rs += __shfl_xor(rs, 1, 32);
      rs += __shfl_xor(rs, 2, 32);
      rs += __shfl_xor(rs, 4, 32);
      rs += __shfl_xor(rs, 8, 32);
      lrow[v] = lrow[v] * alpha + rs;
      mrow[v] = mnew;
#pragma unroll
      for (int nt = 0; nt < 8; nt++) z[nt][v] *= alpha;
      sP[wid][(v + lh) * 32 + n]      = (_Float16)p0;
      sP[wid][(v + lh) * 32 + 16 + n] = (_Float16)p1;
    }

    // ---- z += P (16x32) @ V (32x128), V read from transposed Vt ----
    v16h pa = frag_load(&sP[wid][n * 32 + klo], &sP[wid][n * 32 + klo + 8]);
#pragma unroll
    for (int nt = 0; nt < 8; nt++) {
      int e = nt * 16 + n;
      const _Float16* pv = Vt + (headV + e) * SEQ + kt + kbl;
      z[nt] = WMMA16(pa, frag_load(pv, pv + 8), z[nt]);
    }
  }

  // ---- normalize and store Zh [b][s][h][e] ----
#pragma unroll
  for (int v = 0; v < 8; v++) {
    float inv = 1.0f / lrow[v];
    int s = qlo + v + lh;
#pragma unroll
    for (int nt = 0; nt < 8; nt++) {
      int e = nt * 16 + n;
      Zh[(((size_t)b * SEQ + s) * N_HEADS + h) * D_HEAD + e] =
          (_Float16)(z[nt][v] * inv);
    }
  }
}

// ===========================================================================
// Kernel 3: output projection.  out = Zh(f16) @ W_O + b_O, K = H*Dh = 2048.
// A-fragments direct from global Zh; B tiles (pre-transposed WoT[d][khe])
// DMA'd to LDS by the TDM, double buffered (1 tile/step).
// ===========================================================================
__global__ __launch_bounds__(256)
void oproj_kernel(const _Float16* __restrict__ Zh,
                  const _Float16* __restrict__ WoT,
                  const float* __restrict__ bo,
                  float* __restrict__ out) {
  __shared__ _Float16 sW[2][128 * 32];    // [buf][n*32+k]

  const int tid  = threadIdx.x;
  const int wid  = tid >> 5;
  const int lane = tid & 31;
  const int r0 = blockIdx.x * 128;        // flat b*s rows
  const int n0 = blockIdx.y * 128;        // d_model cols

  v8f acc[8] = {};
  const int klo = (lane < 16) ? 0 : 16;
  const int kbl = (lane < 16) ? 0 : 16;
  const int mr  = r0 + wid * 16 + (lane & 15);

  const _Float16* Wbase = WoT + (size_t)n0 * D_MODEL;   // 128 rows of WoT

  if (wid == 0)
    tdm_load_tile_f16(Wbase, (unsigned)(size_t)&sW[0][0],
                      /*rows*/ 128, /*cols*/ 32,
                      /*tensor_w*/ D_MODEL, /*tensor_h*/ 128,
                      /*stride*/ D_MODEL);

  const int NSTEP = D_MODEL / 32;
  for (int ks = 0; ks < NSTEP; ks++) {
    const int k0  = ks * 32;
    const int cur = ks & 1;
    if (wid == 0) {
      if (ks + 1 < NSTEP) {
        tdm_load_tile_f16(Wbase + (k0 + 32), (unsigned)(size_t)&sW[(ks + 1) & 1][0],
                          128, 32, D_MODEL, 128, D_MODEL);
        __builtin_amdgcn_s_wait_tensorcnt(1);
      } else {
        __builtin_amdgcn_s_wait_tensorcnt(0);
      }
    }
    __syncthreads();

    const _Float16* pa = Zh + (size_t)mr * D_MODEL + k0 + klo;
    v16h a = frag_load(pa, pa + 8);
#pragma unroll
    for (int nt = 0; nt < 8; nt++) {
      int nl = nt * 16 + (lane & 15);
      v16h bf = frag_load(&sW[cur][nl * 32 + kbl], &sW[cur][nl * 32 + kbl + 8]);
      acc[nt] = WMMA16(a, bf, acc[nt]);
    }
    __syncthreads();
  }

  const int lh = (lane < 16) ? 0 : 8;
#pragma unroll
  for (int nt = 0; nt < 8; nt++) {
    int col = n0 + nt * 16 + (lane & 15);
    float bias = bo[col];
#pragma unroll
    for (int v = 0; v < 8; v++) {
      int row = r0 + wid * 16 + v + lh;
      out[(size_t)row * D_MODEL + col] = acc[nt][v] + bias;
    }
  }
}

// ===========================================================================
extern "C" void kernel_launch(void* const* d_in, const int* in_sizes, int n_in,
                              void* d_out, int out_size, void* d_ws, size_t ws_size,
                              hipStream_t stream) {
  (void)in_sizes; (void)n_in; (void)out_size; (void)ws_size;
  const float* x  = (const float*)d_in[0];
  const float* Wq = (const float*)d_in[1];
  const float* Wk = (const float*)d_in[2];
  const float* Wv = (const float*)d_in[3];
  const float* Wo = (const float*)d_in[4];
  const float* bq = (const float*)d_in[5];
  const float* bk = (const float*)d_in[6];
  const float* bv = (const float*)d_in[7];
  const float* bo = (const float*)d_in[8];
  float* out = (float*)d_out;

  // Workspace layout (f16 elements):
  const size_t NX = (size_t)BS * D_MODEL;                  // 8388608  (x, Q/K/V/Z)
  const size_t NW = (size_t)N_HEADS * D_HEAD * D_MODEL;    // 4194304  (each weight)
  _Float16* ws  = (_Float16*)d_ws;
  _Float16* xh  = ws;                  // [bs][d]
  _Float16* WqT = xh  + NX;            // [h][e][d]
  _Float16* WkT = WqT + NW;
  _Float16* WvT = WkT + NW;
  _Float16* WoT = WvT + NW;            // [d][h*e]
  _Float16* Qh  = WoT + NW;            // [b][h][s][e]
  _Float16* Kh  = Qh  + NX;
  _Float16* Vt  = Kh  + NX;            // [b][h][e][s]
  _Float16* Zh  = Vt  + NX;            // [b][s][h*e]

  // ---- precast: x -> f16; weights -> transposed f16 ----
  cast_kernel<<<dim3(NX / 1024), 256, 0, stream>>>(x, xh);
  transpose_cast_kernel<<<dim3(D_HEAD / 32, D_MODEL / 32, N_HEADS), 256, 0, stream>>>(
      Wq, WqT, D_MODEL, D_HEAD);
  transpose_cast_kernel<<<dim3(D_HEAD / 32, D_MODEL / 32, N_HEADS), 256, 0, stream>>>(
      Wk, WkT, D_MODEL, D_HEAD);
  transpose_cast_kernel<<<dim3(D_HEAD / 32, D_MODEL / 32, N_HEADS), 256, 0, stream>>>(
      Wv, WvT, D_MODEL, D_HEAD);
  transpose_cast_kernel<<<dim3(D_MODEL / 32, D_MODEL / 32, 1), 256, 0, stream>>>(
      Wo, WoT, D_MODEL, D_MODEL);

  // ---- QKV projection ----
  qkv_kernel<<<dim3(BS / 128, N_HEADS), 256, 0, stream>>>(
      xh, WqT, WkT, WvT, bq, bk, bv, Qh, Kh, Vt);

  // ---- causal flash attention ----
  flash_kernel<<<dim3(SEQ / 128, N_HEADS, BATCH), 256, 0, stream>>>(Qh, Kh, Vt, Zh);

  // ---- output projection ----
  oproj_kernel<<<dim3(BS / 128, D_MODEL / 128), 256, 0, stream>>>(Zh, WoT, bo, out);
}
